// TrueLiquidTimeConstant_8839042695365
// MI455X (gfx1250) — compile-verified, matrix operand order checked by reading, not run on hardware
//
#include <hip/hip_runtime.h>
#include <cstdint>
#include <cstddef>

#define B_   64
#define T_   256
#define D_   128
#define H_   256
#define UNF_ 6

typedef _Float16 h2    __attribute__((ext_vector_type(2)));
typedef _Float16 v16h  __attribute__((ext_vector_type(16)));
typedef float    v8f   __attribute__((ext_vector_type(8)));

template <typename TO, typename FROM>
static __device__ __forceinline__ TO bcast(FROM f) {
  static_assert(sizeof(TO) == sizeof(FROM), "size mismatch");
  return __builtin_bit_cast(TO, f);
}

static __device__ __forceinline__ float softplus_f(float x) {
  return logf(1.0f + expf(x));
}

// f32 tanh: builtin if declared, else raw V_TANH_F32 (gfx1250 TRANS op).
static __device__ __forceinline__ float tanh_fast(float x) {
#if __has_builtin(__builtin_amdgcn_tanhf)
  return __builtin_amdgcn_tanhf(x);
#elif __has_builtin(__builtin_amdgcn_tanh_f32)
  return __builtin_amdgcn_tanh_f32(x);
#else
  float r;
  asm("v_tanh_f32 %0, %1" : "=v"(r) : "v"(x));
  return r;
#endif
}

// packed-f16 tanh: one V_TANH_F16 per half, operating on .l/.h in place.
static __device__ __forceinline__ h2 tanh_h2(h2 x) {
#if __has_builtin(__builtin_amdgcn_tanh_f16)
  h2 r;
  r.x = __builtin_amdgcn_tanh_f16(x.x);
  r.y = __builtin_amdgcn_tanh_f16(x.y);
  return r;
#else
  h2 r;
  asm("v_tanh_f16 %0.l, %1.l\n\t"
      "v_tanh_f16 %0.h, %1.h"
      : "=v"(r) : "v"(x));
  return r;
#endif
}

static __device__ __forceinline__ uint32_t pk16(float a, float b) {
  return bcast<uint32_t>(__builtin_amdgcn_cvt_pkrtz(a, b));
}

// ---------------------------------------------------------------------------
// Workspace layout (bytes)
// ---------------------------------------------------------------------------
static constexpr size_t OFF_XN    = 0;                         // B*T*D f32  (8 MB)
static constexpr size_t OFF_SN    = 8388608;                   // B*T*H f32  (16 MB)
static constexpr size_t OFF_SD    = 25165824;                  // B*T*H f32  (16 MB)
static constexpr size_t OFF_RECA  = 41943040;                  // H*H/2 dw   (128 KB)
static constexpr size_t OFF_RECB  = OFF_RECA + 131072;         // contiguous with RECA
static constexpr size_t OFF_RECWE = OFF_RECB + 131072;
static constexpr size_t OFF_RECWP = OFF_RECWE + 131072;
static constexpr size_t OFF_SAB   = OFF_RECWP + 131072;        // D*H dw (128 KB)
static constexpr size_t OFF_SWW   = OFF_SAB + 131072;
static constexpr size_t OFF_CMT   = OFF_SWW + 131072;          // H f32
static constexpr size_t OFF_GLK   = OFF_CMT + 1024;
static constexpr size_t OFF_LK    = OFF_GLK + 1024;

// ---------------------------------------------------------------------------
// K1: LayerNorm + input affine.  One wave per row of D=128.
// ---------------------------------------------------------------------------
__global__ __launch_bounds__(256) void ln_kernel(
    const float* __restrict__ x, const float* __restrict__ gamma,
    const float* __restrict__ beta, const float* __restrict__ iw,
    const float* __restrict__ ibias, float* __restrict__ xn) {
  const int lane = threadIdx.x & 31;
  const int wv   = threadIdx.x >> 5;
  const int row  = blockIdx.x * 8 + wv;     // < B*T
  const float* xr = x + (size_t)row * D_;
  float vals[4];
  float s = 0.f, ss = 0.f;
#pragma unroll
  for (int m = 0; m < 4; ++m) {
    float v = xr[lane + 32 * m];
    vals[m] = v; s += v; ss += v * v;
  }
#pragma unroll
  for (int off = 16; off > 0; off >>= 1) {
    s  += __shfl_xor(s, off, 32);
    ss += __shfl_xor(ss, off, 32);
  }
  const float mean = s * (1.0f / D_);
  const float var  = ss * (1.0f / D_) - mean * mean;
  const float inv  = rsqrtf(var + 1e-5f);
#pragma unroll
  for (int m = 0; m < 4; ++m) {
    int d = lane + 32 * m;
    float v = (vals[m] - mean) * inv * gamma[d] + beta[d];
    xn[(size_t)row * D_ + d] = v * iw[d] + ibias[d];
  }
}

// ---------------------------------------------------------------------------
// K2a: recurrent param packing into WMMA B-operand consumption order.
//   p = ((jt*8 + c)*32 + lane)*8 + v ; element pair (i0,i0+1), i0 = c*32 + hi*16 + 2v
// ---------------------------------------------------------------------------
__global__ __launch_bounds__(256) void prep_rec(
    const float* __restrict__ w, const float* __restrict__ mu,
    const float* __restrict__ sigma, const float* __restrict__ erev,
    const float* __restrict__ mask, uint32_t* __restrict__ recA,
    uint32_t* __restrict__ recB, uint32_t* __restrict__ recWE,
    uint32_t* __restrict__ recWP) {
  const int p = blockIdx.x * blockDim.x + threadIdx.x;
  if (p >= H_ * H_ / 2) return;
  const int v    = p & 7;
  const int lane = (p >> 3) & 31;
  const int c    = (p >> 8) & 7;
  const int jt   = p >> 11;
  const int n    = lane & 15;
  const int hi   = lane >> 4;
  const int i0   = c * 32 + hi * 16 + 2 * v;
  const int j    = jt * 16 + n;

  float a[2], bt[2], hwe[2], hwp[2];
#pragma unroll
  for (int e = 0; e < 2; ++e) {
    const int idx = (i0 + e) * H_ + j;
    float sg = sigma[idx], m = mu[idx];
    float wp = softplus_f(w[idx]) * mask[idx];
    float we = wp * erev[idx];
    a[e]   = 0.5f * sg;
    bt[e]  = -0.5f * sg * m;
    hwe[e] = 0.5f * we;
    hwp[e] = 0.5f * wp;
  }
  recA[p]  = pk16(a[0], a[1]);
  recB[p]  = pk16(bt[0], bt[1]);
  recWE[p] = pk16(hwe[0], hwe[1]);
  recWP[p] = pk16(hwp[0], hwp[1]);
}

// ---------------------------------------------------------------------------
// K2b: sensory param packing, (d,j) linear order.
// ---------------------------------------------------------------------------
__global__ __launch_bounds__(256) void prep_sens(
    const float* __restrict__ sw, const float* __restrict__ smu,
    const float* __restrict__ ssig, const float* __restrict__ serev,
    const float* __restrict__ smask, uint32_t* __restrict__ sAB,
    uint32_t* __restrict__ sWW) {
  const int p = blockIdx.x * blockDim.x + threadIdx.x;
  if (p >= D_ * H_) return;
  float sg = ssig[p], m = smu[p];
  float wp = softplus_f(sw[p]) * smask[p];
  float we = wp * serev[p];
  sAB[p] = pk16(0.5f * sg, -0.5f * sg * m);
  sWW[p] = pk16(0.5f * we, 0.5f * wp);
}

// ---------------------------------------------------------------------------
// K2c: per-neuron vectors.
// ---------------------------------------------------------------------------
__global__ __launch_bounds__(256) void prep_vec(
    const float* __restrict__ gleak, const float* __restrict__ vleak,
    const float* __restrict__ cm, float* __restrict__ cmt,
    float* __restrict__ glk, float* __restrict__ lk) {
  const int j = threadIdx.x;
  float gp = softplus_f(gleak[j]);
  cmt[j] = softplus_f(cm[j]) * ((float)UNF_ * 30.0f);  // softplus(cm)/(DT/UNF)
  glk[j] = gp;
  lk[j]  = gp * vleak[j];
}

// ---------------------------------------------------------------------------
// K3: sensory sums for all (b,t): sn/sd[b,t,j] = sum_d we*s / wp*s.
// One block per (b, 32-t tile); params staged in 256 KB LDS.
// ---------------------------------------------------------------------------
__global__ __launch_bounds__(256) void sens_kernel(
    const float* __restrict__ xn, const uint32_t* __restrict__ sABg,
    const uint32_t* __restrict__ sWWg, float* __restrict__ sn,
    float* __restrict__ sd) {
  extern __shared__ char sm3[];
  uint32_t* ab  = (uint32_t*)sm3;          // D*H
  uint32_t* wwp = ab + D_ * H_;            // D*H
  float*    xl  = (float*)(wwp + D_ * H_); // D
  const int tid = threadIdx.x;
  for (int r = tid; r < D_ * H_; r += 256) { ab[r] = sABg[r]; wwp[r] = sWWg[r]; }
  __syncthreads();

  const int b  = blockIdx.x >> 3;
  const int t0 = (blockIdx.x & 7) * 32;
  const int j  = tid;
  for (int tt = 0; tt < 32; ++tt) {
    const int t = t0 + tt;
    if (tid < D_) xl[tid] = xn[((size_t)(b * T_ + t)) * D_ + tid];
    __syncthreads();
    float an = 0.f, ad = 0.f;
    for (int d = 0; d < D_; ++d) {
      h2 abp = bcast<h2>(ab[d * H_ + j]);
      h2 wwq = bcast<h2>(wwp[d * H_ + j]);
      float arg = fmaf((float)abp.x, xl[d], (float)abp.y);  // 0.5*sig*(x-mu)
      float tp1 = tanh_fast(arg) + 1.0f;                    // 2*sigmoid(...)
      an = fmaf((float)wwq.x, tp1, an);                     // += we*s
      ad = fmaf((float)wwq.y, tp1, ad);                     // += wp*s
    }
    sn[((size_t)(b * T_ + t)) * H_ + j] = an;
    sd[((size_t)(b * T_ + t)) * H_ + j] = ad;
    __syncthreads();
  }
}

// ---------------------------------------------------------------------------
// K4: persistent recurrent kernel. 1 workgroup per batch, 512 threads = 16 waves.
// Wave jt owns columns [jt*16, jt*16+16). Per unfold: 8 K-chunks of 32 rows,
// reduction over rows via v_wmma_f32_16x16x32_f16 with A = ones.
// ---------------------------------------------------------------------------
union BOp { h2 p[8]; v16h v; };

__global__ __launch_bounds__(512, 1) void ltc_kernel(
    const uint32_t* __restrict__ pABg,   // recA|recB contiguous, 65536 dwords
    const uint32_t* __restrict__ pWEg,   // 32768 dwords
    const uint32_t* __restrict__ pWPg,   // 32768 dwords
    const float* __restrict__ cmtv, const float* __restrict__ glkv,
    const float* __restrict__ lkv, const float* __restrict__ snv,
    const float* __restrict__ sdv, float* __restrict__ out) {
  extern __shared__ char smem[];
  uint32_t* pAB = (uint32_t*)smem;                  // 65536 dwords (256 KB)
  float*    vb  = (float*)(smem + 65536 * 4);       // 2 * H state ping-pong

  const int tid  = threadIdx.x;
  const int lane = tid & 31;
  const int wv   = tid >> 5;        // jt, 0..15
  const int n    = lane & 15;
  const int hi   = lane >> 4;
  const int j    = wv * 16 + n;
  const int b    = blockIdx.x;

  // ---- stage 256 KB of packed (a, b) sigmoid params into LDS via TDM ----
#if __has_builtin(__builtin_amdgcn_tensor_load_to_lds)
  if (wv == 0) {
    typedef unsigned int v4u __attribute__((ext_vector_type(4)));
    typedef int v8i __attribute__((ext_vector_type(8)));
    typedef int v4i __attribute__((ext_vector_type(4)));
    unsigned long long ga = (unsigned long long)(uintptr_t)pABg;
    unsigned int lds0 = (unsigned int)(uintptr_t)(void*)smem;
    v4u g0;
    g0[0] = 1u;                                            // count=1, user mode
    g0[1] = lds0;                                          // lds_addr
    g0[2] = (unsigned int)(ga & 0xffffffffull);            // global_addr lo
    g0[3] = (unsigned int)((ga >> 32) & 0x01ffffffull) | (2u << 30);  // hi + type=2
    v8i g1;
    g1[0] = (int)(2u << 16);       // data_size = 4B
    g1[1] = (int)(32768u << 16);   // tensor_dim0 = 32768 dwords (lo16 -> [31:16])
    g1[2] = (int)(2u << 16);       // tensor_dim0 hi = 0; tensor_dim1 = 2
    g1[3] = (int)(32768u << 16);   // tensor_dim1 hi = 0; tile_dim0 = 32768
    g1[4] = (int)2;                // tile_dim1 = 2, tile_dim2 = 0
    g1[5] = (int)32768;            // tensor_dim0_stride = 32768 (contiguous rows)
    g1[6] = (int)(32768u << 16);   // stride0 hi = 0; tensor_dim1_stride lo16
    g1[7] = 0;                     // tensor_dim1_stride hi = 0
    v4i g2 = (v4i)0, g3 = (v4i)0;  // groups 2/3 unused for 2D tile
    v8i g4 = (v8i)0;               // extra operand (clang-23 6-arg form), zeros
    __builtin_amdgcn_tensor_load_to_lds(g0, g1, g2, g3, g4, 0);
#if __has_builtin(__builtin_amdgcn_s_wait_tensorcnt)
    __builtin_amdgcn_s_wait_tensorcnt(0);
#endif
  }
#else
  for (int r = tid; r < 65536; r += 512) pAB[r] = pABg[r];
#endif
  if (tid < H_) { vb[tid] = 0.0f; vb[H_ + tid] = 0.0f; }
  __syncthreads();

  float cmt = 0.f, glk = 0.f, lk = 0.f;
  if (hi == 0) { cmt = cmtv[j]; glk = glkv[j]; lk = lkv[j]; }

  v16h aone;
#pragma unroll
  for (int e = 0; e < 16; ++e) aone[e] = (_Float16)1.0f;

  float vreg = 0.0f;
  int p = 0;
  for (int t = 0; t < T_; ++t) {
    float sn_t = 0.f, sd_t = 0.f;
    if (hi == 0) {
      size_t o = ((size_t)(b * T_ + t)) * H_ + j;
      sn_t = snv[o];
      sd_t = sdv[o];
    }
    for (int u = 0; u < UNF_; ++u) {
      v8f cn = {}, cd = {};
      const float* vcur = vb + p * H_;
      for (int c = 0; c < 8; ++c) {
        const int p4 = (((wv * 8 + c) * 32) + lane) * 2;   // uint4 index
        const uint4* lds4 = (const uint4*)pAB;
        uint4 A0 = lds4[p4],        A1 = lds4[p4 + 1];
        uint4 B0 = lds4[8192 + p4], B1 = lds4[8192 + p4 + 1];
        uint4 E0 = ((const uint4*)pWEg)[p4], E1 = ((const uint4*)pWEg)[p4 + 1];
        uint4 P0 = ((const uint4*)pWPg)[p4], P1 = ((const uint4*)pWPg)[p4 + 1];
        const float4* v4 = (const float4*)(vcur + c * 32 + hi * 16);
        float4 q0 = v4[0], q1 = v4[1], q2 = v4[2], q3 = v4[3];
        uint32_t aA[8] = {A0.x, A0.y, A0.z, A0.w, A1.x, A1.y, A1.z, A1.w};
        uint32_t aB[8] = {B0.x, B0.y, B0.z, B0.w, B1.x, B1.y, B1.z, B1.w};
        uint32_t aE[8] = {E0.x, E0.y, E0.z, E0.w, E1.x, E1.y, E1.z, E1.w};
        uint32_t aP[8] = {P0.x, P0.y, P0.z, P0.w, P1.x, P1.y, P1.z, P1.w};
        float4 q[4] = {q0, q1, q2, q3};
        BOp bn, bd;
#pragma unroll
        for (int v = 0; v < 8; ++v) {
          float v0 = (v & 1) ? q[v >> 1].z : q[v >> 1].x;
          float v1 = (v & 1) ? q[v >> 1].w : q[v >> 1].y;
          h2 vp  = bcast<h2>(__builtin_amdgcn_cvt_pkrtz(v0, v1));
          h2 Ai  = bcast<h2>(aA[v]);
          h2 Bi  = bcast<h2>(aB[v]);
          h2 arg = Ai * vp + Bi;              // v_pk_fma_f16
          h2 th  = tanh_h2(arg);              // tanh = 2*sigmoid-1
          h2 We  = bcast<h2>(aE[v]);
          h2 Wp  = bcast<h2>(aP[v]);
          bn.p[v] = We * th + We;             // 0.5we*(t+1) = we*sigmoid
          bd.p[v] = Wp * th + Wp;
        }
        // column sums over this 32-row chunk, accumulated in f32 C
        cn = __builtin_amdgcn_wmma_f32_16x16x32_f16(false, aone, false, bn.v,
                                                    (short)0, cn, false, false);
        cd = __builtin_amdgcn_wmma_f32_16x16x32_f16(false, aone, false, bd.v,
                                                    (short)0, cd, false, false);
      }
      float* vnext = vb + (1 - p) * H_;
      if (hi == 0) {
        float wn = cn[0] + sn_t;
        float wd = cd[0] + sd_t;
        float vnew = (cmt * vreg + lk + wn) / (cmt + glk + wd + 1e-8f);
        vreg = vnew;
        vnext[j] = vnew;
      }
      __syncthreads();
      p = 1 - p;
    }
    if (hi == 0) out[((size_t)(b * T_ + t)) * H_ + j] = vreg;
  }
  if (hi == 0) {
    out[(size_t)B_ * T_ * H_ + (size_t)b * H_ + j] = vreg;
    out[(size_t)B_ * T_ * H_ + (size_t)B_ * H_ + (size_t)b * H_ + j] = vreg;
  }
}

// ---------------------------------------------------------------------------
extern "C" void kernel_launch(void* const* d_in, const int* in_sizes, int n_in,
                              void* d_out, int out_size, void* d_ws,
                              size_t ws_size, hipStream_t stream) {
  const float* x     = (const float*)d_in[0];
  const float* w     = (const float*)d_in[1];
  const float* mu    = (const float*)d_in[2];
  const float* sigma = (const float*)d_in[3];
  const float* erev  = (const float*)d_in[4];
  const float* mask  = (const float*)d_in[5];
  const float* sw    = (const float*)d_in[6];
  const float* smu   = (const float*)d_in[7];
  const float* ssig  = (const float*)d_in[8];
  const float* serev = (const float*)d_in[9];
  const float* smask = (const float*)d_in[10];
  const float* gleak = (const float*)d_in[11];
  const float* vleak = (const float*)d_in[12];
  const float* cm    = (const float*)d_in[13];
  const float* iw    = (const float*)d_in[14];
  const float* ibias = (const float*)d_in[15];
  const float* lng   = (const float*)d_in[16];
  const float* lnb   = (const float*)d_in[17];

  char* ws = (char*)d_ws;
  float*    xn    = (float*)(ws + OFF_XN);
  float*    sn    = (float*)(ws + OFF_SN);
  float*    sd    = (float*)(ws + OFF_SD);
  uint32_t* recA  = (uint32_t*)(ws + OFF_RECA);
  uint32_t* recB  = (uint32_t*)(ws + OFF_RECB);
  uint32_t* recWE = (uint32_t*)(ws + OFF_RECWE);
  uint32_t* recWP = (uint32_t*)(ws + OFF_RECWP);
  uint32_t* sAB   = (uint32_t*)(ws + OFF_SAB);
  uint32_t* sWW   = (uint32_t*)(ws + OFF_SWW);
  float*    cmt   = (float*)(ws + OFF_CMT);
  float*    glk   = (float*)(ws + OFF_GLK);
  float*    lk    = (float*)(ws + OFF_LK);

  ln_kernel<<<(B_ * T_) / 8, 256, 0, stream>>>(x, lng, lnb, iw, ibias, xn);
  prep_rec<<<(H_ * H_ / 2) / 256, 256, 0, stream>>>(w, mu, sigma, erev, mask,
                                                    recA, recB, recWE, recWP);
  prep_sens<<<(D_ * H_) / 256, 256, 0, stream>>>(sw, smu, ssig, serev, smask,
                                                 sAB, sWW);
  prep_vec<<<1, 256, 0, stream>>>(gleak, vleak, cm, cmt, glk, lk);

  size_t lds3 = (size_t)(2 * D_ * H_ * 4 + D_ * 4);       // 262656 B
  sens_kernel<<<B_ * (T_ / 32), 256, lds3, stream>>>(xn, sAB, sWW, sn, sd);

  size_t lds4 = (size_t)(65536 * 4 + 2 * H_ * 4);         // 264192 B
  ltc_kernel<<<B_, 512, lds4, stream>>>(recA, recWE, recWP, cmt, glk, lk, sn,
                                        sd, (float*)d_out);
}
